// ENC_GRUTurbo_52527450030498
// MI455X (gfx1250) — matrix operand order repaired and checked by reading, not compiled, not run on hardware
//
#include <hip/hip_runtime.h>
#include <hip/hip_bf16.h>

// ---------------------------------------------------------------------------
// minGRU stacked encoder for MI455X (gfx1250), wave32 + WMMA bf16.
// B=256, L=2048, H=64, M=B*L=524288.
// Pipeline:
//   k1 layer0_scan  : binary-input minGRU scan (both branches) -> h0 (bf16)
//   k2 gemm_gate    : h0 @ {Wz1,Wh1} via v_wmma_f32_16x16x32_bf16, fused
//                     sigmoid gate epilogue -> packed (a|b) bf16x2 per u32
//   k3 scan1        : sequential scan over packed gates -> h1 (bf16, in-place
//                     over h0 buffer)
//   k4 project      : h1 @ Wo + bias for both branches -> out(B,L,2) + block
//                     partial sums for mean/var
//   k5 finalize     : deterministic reduction -> mean, 1/std (ddof=1)
//   k6 normalize    : out = (out - mean) * invstd
// ---------------------------------------------------------------------------

#define BQ 256
#define LQ 2048
#define HQ 64
#define MQ (BQ * LQ)          // 524288 rows

typedef __attribute__((ext_vector_type(16))) __bf16 v16bf;
typedef __attribute__((ext_vector_type(8)))  float  v8f;

union FragU { uint4 q[2]; v16bf v; };

__device__ __forceinline__ unsigned short f2bf(float f) {
    unsigned u = __float_as_uint(f);
    unsigned r = u + 0x7FFFu + ((u >> 16) & 1u);   // round-to-nearest-even
    return (unsigned short)(r >> 16);
}
__device__ __forceinline__ float bf_lo(unsigned u) { return __uint_as_float(u << 16); }
__device__ __forceinline__ float bf_hi(unsigned u) { return __uint_as_float(u & 0xFFFF0000u); }
__device__ __forceinline__ float sigmoidf(float x) { return 1.0f / (1.0f + __expf(-x)); }

// ---------------------------------------------------------------------------
// k1: layer-0 minGRU. x is {0,1}, feature dim 1 -> only two (a,b) pairs per
// channel. One block per batch element; threads 0..63 = sys channels,
// 64..127 = permuted-branch channels. x row + permutation staged in LDS.
// ---------------------------------------------------------------------------
__global__ __launch_bounds__(128) void layer0_scan(
    const float* __restrict__ x, const int* __restrict__ p,
    const float* __restrict__ wz_s, const float* __restrict__ bz_s,
    const float* __restrict__ wh_s, const float* __restrict__ bh_s,
    const float* __restrict__ wz_p, const float* __restrict__ bz_p,
    const float* __restrict__ wh_p, const float* __restrict__ bh_p,
    unsigned short* __restrict__ A0s, unsigned short* __restrict__ A0p)
{
    __shared__ float xs[LQ];
    __shared__ int   ps[LQ];
    const int b   = blockIdx.x;
    const int tid = threadIdx.x;

    for (int t = tid; t < LQ; t += 128) {
        xs[t] = x[(size_t)b * LQ + t];
        ps[t] = p[t];
    }
    __syncthreads();

    const int  h    = tid & 63;
    const bool perm = (tid >= 64);
    const float wzv = perm ? wz_p[h] : wz_s[h];
    const float bzv = perm ? bz_p[h] : bz_s[h];
    const float whv = perm ? wh_p[h] : wh_s[h];
    const float bhv = perm ? bh_p[h] : bh_s[h];

    const float z0 = sigmoidf(bzv);
    const float z1 = sigmoidf(bzv + wzv);
    const float a0 = 1.0f - z0, b0 = z0 * bhv;
    const float a1 = 1.0f - z1, b1 = z1 * (bhv + whv);

    unsigned short* out = (perm ? A0p : A0s) + (size_t)b * LQ * HQ + h;
    float hs = 0.0f;
    for (int t = 0; t < LQ; ++t) {
        const float xv  = perm ? xs[ps[t]] : xs[t];
        const bool  one = (xv != 0.0f);
        const float a   = one ? a1 : a0;
        const float bb  = one ? b1 : b0;
        hs = fmaf(a, hs, bb);
        out[(size_t)t * HQ] = f2bf(hs);
    }
}

// ---------------------------------------------------------------------------
// k2: layer-1 gates via WMMA. Each wave computes one 16x64 tile of both
// Z = A@Wz and H = A@Wh, then fuses a = 1-sigmoid(Z+bz), b = sigmoid*(H+bh)
// and stores packed bf16x2. Weights staged transposed (N-major) in LDS with
// a 72-element row stride (16B-aligned, bank-conflict friendly).
// ---------------------------------------------------------------------------
#define WSTRIDE 72

__global__ __launch_bounds__(256) void gemm_gate(
    const unsigned short* __restrict__ A,   // M x 64 bf16 row-major
    const float* __restrict__ Wz, const float* __restrict__ bz,
    const float* __restrict__ Wh, const float* __restrict__ bh,
    unsigned* __restrict__ AB)              // M x 64 packed (a|b)
{
    __shared__ unsigned short WTz[HQ * WSTRIDE];
    __shared__ unsigned short WTh[HQ * WSTRIDE];

    const int tid = threadIdx.x;
    for (int e = tid; e < HQ * HQ; e += 256) {
        const int i = e >> 6;        // input (K) index
        const int o = e & 63;        // output (N) index
        WTz[o * WSTRIDE + i] = f2bf(Wz[e]);
        WTh[o * WSTRIDE + i] = f2bf(Wh[e]);
    }
    __syncthreads();

    const int lane = tid & 31;
    const int wave = tid >> 5;
    const int m0   = (blockIdx.x * 8 + wave) * 16;

    // A fragments (16x32 bf16, ISA layout): lanes 0-15 hold row m0+lane,
    // K in {kb..kb+7, 16+kb..16+kb+7} with kb = 8*(lane>>4).
    const int row = m0 + (lane & 15);
    const int kb  = (lane >> 4) * 8;
    const unsigned short* arow = A + (size_t)row * HQ;
    __builtin_prefetch(arow + (size_t)128 * HQ, 0, 0);   // next block's tile

    FragU a0, a1;
    a0.q[0] = *(const uint4*)(arow + kb);
    a0.q[1] = *(const uint4*)(arow + 16 + kb);
    a1.q[0] = *(const uint4*)(arow + 32 + kb);
    a1.q[1] = *(const uint4*)(arow + 48 + kb);

    const int ncol  = lane & 15;           // N within 16-wide tile
    const int koff  = (lane >> 4) * 8;     // B-fragment K base (table order)
    const int rbase = m0 + ((lane >> 4) << 3);

    for (int nt = 0; nt < 4; ++nt) {
        const int col = nt * 16 + ncol;
        v8f accz = {};
        v8f acch = {};

        FragU bzf, bhf;
        // k-chunk 0 : K in {koff..koff+7, 16+koff..16+koff+7}
        bzf.q[0] = *(const uint4*)(&WTz[col * WSTRIDE + koff]);
        bzf.q[1] = *(const uint4*)(&WTz[col * WSTRIDE + koff + 16]);
        bhf.q[0] = *(const uint4*)(&WTh[col * WSTRIDE + koff]);
        bhf.q[1] = *(const uint4*)(&WTh[col * WSTRIDE + koff + 16]);
        accz = __builtin_amdgcn_wmma_f32_16x16x32_bf16(false, a0.v, false, bzf.v,
                                                       (short)0, accz, false, false);
        acch = __builtin_amdgcn_wmma_f32_16x16x32_bf16(false, a0.v, false, bhf.v,
                                                       (short)0, acch, false, false);
        // k-chunk 1 : K = 32 + koff ..
        bzf.q[0] = *(const uint4*)(&WTz[col * WSTRIDE + 32 + koff]);
        bzf.q[1] = *(const uint4*)(&WTz[col * WSTRIDE + 32 + koff + 16]);
        bhf.q[0] = *(const uint4*)(&WTh[col * WSTRIDE + 32 + koff]);
        bhf.q[1] = *(const uint4*)(&WTh[col * WSTRIDE + 32 + koff + 16]);
        accz = __builtin_amdgcn_wmma_f32_16x16x32_bf16(false, a1.v, false, bzf.v,
                                                       (short)0, accz, false, false);
        acch = __builtin_amdgcn_wmma_f32_16x16x32_bf16(false, a1.v, false, bhf.v,
                                                       (short)0, acch, false, false);

        const float bzc = bz[col];
        const float bhc = bh[col];
#pragma unroll
        for (int v = 0; v < 8; ++v) {
            const float z  = sigmoidf(accz[v] + bzc);
            const float ht = acch[v] + bhc;
            const float av = 1.0f - z;
            const float bv = z * ht;
            AB[(size_t)(rbase + v) * HQ + col] =
                (unsigned)f2bf(av) | ((unsigned)f2bf(bv) << 16);
        }
    }
}

// ---------------------------------------------------------------------------
// k3: layer-1 scan. One thread per (b,h) channel; lane = h -> coalesced u32
// stream per time step. h1 overwrites the consumed h0 buffer.
// ---------------------------------------------------------------------------
__global__ __launch_bounds__(256) void scan1(
    const unsigned* __restrict__ AB, unsigned short* __restrict__ H1)
{
    const int gid = blockIdx.x * 256 + threadIdx.x;   // 0 .. 16383
    const int b = gid >> 6;
    const int h = gid & 63;
    const unsigned*  in  = AB + (size_t)b * LQ * HQ + h;
    unsigned short*  out = H1 + (size_t)b * LQ * HQ + h;
    float hs = 0.0f;
    for (int t = 0; t < LQ; ++t) {
        const unsigned u = in[(size_t)t * HQ];
        hs = fmaf(bf_lo(u), hs, bf_hi(u));
        out[(size_t)t * HQ] = f2bf(hs);
    }
}

// ---------------------------------------------------------------------------
// k4: output projection (H -> 1) for both branches + block partial sums for
// the global layernorm (deterministic: LDS tree reduction, no float atomics).
// ---------------------------------------------------------------------------
__global__ __launch_bounds__(256) void project(
    const unsigned short* __restrict__ H1s, const unsigned short* __restrict__ H1p,
    const float* __restrict__ wo1, const float* __restrict__ bo1,
    const float* __restrict__ wo2, const float* __restrict__ bo2,
    float* __restrict__ out, float2* __restrict__ partials)
{
    __shared__ float w1s[HQ], w2s[HQ];
    __shared__ float rs[256], rss[256];
    const int tid = threadIdx.x;
    if (tid < HQ) { w1s[tid] = wo1[tid]; w2s[tid] = wo2[tid]; }
    __syncthreads();

    const size_t rowi = (size_t)blockIdx.x * 256 + tid;   // < 524288
    const unsigned short* r1 = H1s + rowi * HQ;
    const unsigned short* r2 = H1p + rowi * HQ;

    float s1 = 0.0f, s2 = 0.0f;
#pragma unroll
    for (int q = 0; q < 8; ++q) {
        const uint4 u1 = *(const uint4*)(r1 + q * 8);
        const uint4 u2 = *(const uint4*)(r2 + q * 8);
        const unsigned ua[4] = {u1.x, u1.y, u1.z, u1.w};
        const unsigned ub[4] = {u2.x, u2.y, u2.z, u2.w};
#pragma unroll
        for (int j = 0; j < 4; ++j) {
            const int h = q * 8 + j * 2;
            s1 = fmaf(bf_lo(ua[j]), w1s[h],     s1);
            s1 = fmaf(bf_hi(ua[j]), w1s[h + 1], s1);
            s2 = fmaf(bf_lo(ub[j]), w2s[h],     s2);
            s2 = fmaf(bf_hi(ub[j]), w2s[h + 1], s2);
        }
    }
    const float o1 = s1 + bo1[0];
    const float o2 = s2 + bo2[0];
    out[rowi * 2]     = o1;
    out[rowi * 2 + 1] = o2;

    rs[tid]  = o1 + o2;
    rss[tid] = o1 * o1 + o2 * o2;
    __syncthreads();
    for (int s = 128; s > 0; s >>= 1) {
        if (tid < s) { rs[tid] += rs[tid + s]; rss[tid] += rss[tid + s]; }
        __syncthreads();
    }
    if (tid == 0) partials[blockIdx.x] = make_float2(rs[0], rss[0]);
}

// ---------------------------------------------------------------------------
// k5: finalize mean / 1/std (ddof=1) from 2048 block partials.
// ---------------------------------------------------------------------------
__global__ __launch_bounds__(256) void finalize(
    const float2* __restrict__ partials, int n, float* __restrict__ stats)
{
    __shared__ float rs[256], rss[256];
    const int tid = threadIdx.x;
    float s = 0.0f, ss = 0.0f;
    for (int i = tid; i < n; i += 256) {
        const float2 pv = partials[i];
        s += pv.x; ss += pv.y;
    }
    rs[tid] = s; rss[tid] = ss;
    __syncthreads();
    for (int k = 128; k > 0; k >>= 1) {
        if (tid < k) { rs[tid] += rs[tid + k]; rss[tid] += rss[tid + k]; }
        __syncthreads();
    }
    if (tid == 0) {
        const float N    = (float)(MQ * 2);
        const float mean = rs[0] / N;
        const float var  = (rss[0] - rs[0] * rs[0] / N) / (N - 1.0f);
        stats[0] = mean;
        stats[1] = 1.0f / sqrtf(var);
    }
}

// ---------------------------------------------------------------------------
// k6: apply global normalization in place on d_out.
// ---------------------------------------------------------------------------
__global__ __launch_bounds__(512) void normalize(
    float* __restrict__ out, const float* __restrict__ stats, int n)
{
    const int i = blockIdx.x * 512 + threadIdx.x;
    if (i < n) out[i] = (out[i] - stats[0]) * stats[1];
}

// ---------------------------------------------------------------------------
extern "C" void kernel_launch(void* const* d_in, const int* in_sizes, int n_in,
                              void* d_out, int out_size, void* d_ws, size_t ws_size,
                              hipStream_t stream)
{
    (void)in_sizes; (void)n_in; (void)out_size; (void)ws_size;

    const float* x    = (const float*)d_in[0];
    const int*   p    = (const int*)d_in[1];
    const float* w1z0 = (const float*)d_in[2];  const float* b1z0 = (const float*)d_in[3];
    const float* w1h0 = (const float*)d_in[4];  const float* b1h0 = (const float*)d_in[5];
    const float* w1z1 = (const float*)d_in[6];  const float* b1z1 = (const float*)d_in[7];
    const float* w1h1 = (const float*)d_in[8];  const float* b1h1 = (const float*)d_in[9];
    const float* w1o  = (const float*)d_in[10]; const float* b1o  = (const float*)d_in[11];
    const float* w2z0 = (const float*)d_in[12]; const float* b2z0 = (const float*)d_in[13];
    const float* w2h0 = (const float*)d_in[14]; const float* b2h0 = (const float*)d_in[15];
    const float* w2z1 = (const float*)d_in[16]; const float* b2z1 = (const float*)d_in[17];
    const float* w2h1 = (const float*)d_in[18]; const float* b2h1 = (const float*)d_in[19];
    const float* w2o  = (const float*)d_in[20]; const float* b2o  = (const float*)d_in[21];

    float* out = (float*)d_out;
    char*  ws  = (char*)d_ws;

    const size_t szA0 = (size_t)MQ * HQ * sizeof(unsigned short);   // 64 MB
    const size_t szAB = (size_t)MQ * HQ * sizeof(unsigned);         // 128 MB

    unsigned short* A0s = (unsigned short*)(ws);
    unsigned short* A0p = (unsigned short*)(ws + szA0);
    unsigned*       AB  = (unsigned*)(ws + 2 * szA0);
    float2*         prt = (float2*)(ws + 2 * szA0 + szAB);
    float*          sts = (float*)(ws + 2 * szA0 + szAB + 4096 * sizeof(float2));

    // k1: layer-0 scans (both branches)
    layer0_scan<<<BQ, 128, 0, stream>>>(x, p,
                                        w1z0, b1z0, w1h0, b1h0,
                                        w2z0, b2z0, w2h0, b2h0,
                                        A0s, A0p);
    // sys branch: WMMA gates then scan (h1 overwrites A0s)
    gemm_gate<<<MQ / (16 * 8), 256, 0, stream>>>(A0s, w1z1, b1z1, w1h1, b1h1, AB);
    scan1<<<(BQ * HQ) / 256, 256, 0, stream>>>(AB, A0s);
    // permuted branch (reuses AB scratch; stream order guarantees safety)
    gemm_gate<<<MQ / (16 * 8), 256, 0, stream>>>(A0p, w2z1, b2z1, w2h1, b2h1, AB);
    scan1<<<(BQ * HQ) / 256, 256, 0, stream>>>(AB, A0p);
    // projection + global layernorm
    project<<<MQ / 256, 256, 0, stream>>>(A0s, A0p, w1o, b1o, w2o, b2o, out, prt);
    finalize<<<1, 256, 0, stream>>>(prt, MQ / 256, sts);
    normalize<<<(MQ * 2) / 512, 512, 0, stream>>>(out, sts, MQ * 2);
}
